// Model_58600533787270
// MI455X (gfx1250) — compile-verified, matrix operand order checked by reading, not run on hardware
//
#include <hip/hip_runtime.h>
#include <hip/hip_bf16.h>

typedef __attribute__((ext_vector_type(16))) __bf16 v16bf;
typedef __attribute__((ext_vector_type(8)))  float  v8f;
typedef __attribute__((ext_vector_type(4)))  unsigned int v4u;
typedef __attribute__((ext_vector_type(8)))  int v8i;
typedef __attribute__((ext_vector_type(4)))  int v4i;
typedef unsigned short u16;
typedef unsigned int   u32;

// ---------- helpers ----------
__device__ __forceinline__ u16 f2bf(float f) {
  union { float f; u32 u; } c; c.f = f;
  u32 r = c.u + 0x7fffu + ((c.u >> 16) & 1u);   // round-to-nearest-even
  return (u16)(r >> 16);
}
__device__ __forceinline__ float bf2f(u16 u) {
  union { u32 u; float f; } c; c.u = ((u32)u) << 16;
  return c.f;
}
__device__ __forceinline__ v8f vzero() {
  v8f z;
  #pragma unroll
  for (int i = 0; i < 8; ++i) z[i] = 0.f;
  return z;
}
// 16-bf16-per-lane fragment: two contiguous 8-element chunks 16 elements apart
// (CDNA5 16-bit A/B VGPR layout).
__device__ __forceinline__ v16bf load_frag(const u16* p) {
  union { v16bf v; uint4 q[2]; } f;
  f.q[0] = *(const uint4*)(p);
  f.q[1] = *(const uint4*)(p + 16);
  return f.v;
}
__device__ __forceinline__ v8f wmma_bf16(v16bf a, v16bf b, v8f c) {
  return __builtin_amdgcn_wmma_f32_16x16x32_bf16(false, a, false, b, (short)0, c,
                                                 false, false);
}

// 1-D Tensor-Data-Mover copy: global -> LDS, n8 * 8 bytes, lds byte offset lds_off.
__device__ __forceinline__ void tdm_load_1d(u32 lds_off, const void* gaddr, u32 n8) {
  unsigned long long ga = (unsigned long long)gaddr;
  v4u g0;
  g0.x = 0x1u;                                     // count = 1
  g0.y = lds_off;                                  // lds_addr (bytes)
  g0.z = (u32)ga;                                  // global_addr[31:0]
  g0.w = (u32)(ga >> 32) | 0x80000000u;            // global_addr[56:32] | type=2
  v8i g1;
  g1[0] = (int)(3u << 16);                         // data_size = 8 bytes
  g1[1] = (int)((n8 & 0xFFFFu) << 16);             // tensor_dim0 low
  g1[2] = (int)((n8 >> 16) | (1u << 16));          // tensor_dim0 hi | tensor_dim1 = 1
  g1[3] = (int)(n8 << 16);                         // tile_dim0 (16-bit)
  g1[4] = 0;
  g1[5] = (int)n8;                                 // tensor_dim0_stride low
  g1[6] = 0;
  g1[7] = 0;
  v4i z4 = {};
  v8i z8 = {};
  __builtin_amdgcn_tensor_load_to_lds(g0, g1, z4, z4, z8, 0);  // 6-arg (clang-23) form
}

// ---------- sizes ----------
#define MTOT 16384   // B*S
#define DIN  512
#define DMOD 1024
#define VSZ  16384
#define DOUT 512
#define VT   256     // V tile per fused iteration
#define MT   32      // rows per block in fused kernel

// ---------- prep: cast weights ----------
__global__ void prep_w(const float* __restrict__ encw, const float* __restrict__ decw,
                       u16* __restrict__ encw_bf, u16* __restrict__ decw_bf, int n) {
  for (int i = blockIdx.x * blockDim.x + threadIdx.x; i < n; i += gridDim.x * blockDim.x) {
    encw_bf[i] = f2bf(encw[i]);
    decw_bf[i] = f2bf(decw[i]);
  }
}

// ---------- prep: cast + transpose embedding ----------
__global__ void prep_emb(const float* __restrict__ emb,
                         u16* __restrict__ emb_bf, u16* __restrict__ embT_bf) {
  __shared__ float tile[32][33];
  const int tx = threadIdx.x, ty = threadIdx.y;
  const int d0 = blockIdx.x * 32, v0 = blockIdx.y * 32;
  #pragma unroll
  for (int r = 0; r < 4; ++r) {
    int v = v0 + ty + r * 8;
    float val = emb[(size_t)v * DMOD + d0 + tx];
    tile[ty + r * 8][tx] = val;
    emb_bf[(size_t)v * DMOD + d0 + tx] = f2bf(val);
  }
  __syncthreads();
  #pragma unroll
  for (int r = 0; r < 4; ++r) {
    int d = d0 + ty + r * 8;
    embT_bf[(size_t)d * VSZ + v0 + tx] = f2bf(tile[tx][ty + r * 8]);
  }
}

// ---------- encoder GEMM + layernorm (16 rows/block), triple-buffered ----------
__global__ __launch_bounds__(256) void encode_ln(
    const float* __restrict__ x, const u16* __restrict__ encw_bf,
    const float* __restrict__ enc_b, const float* __restrict__ ln_g,
    const float* __restrict__ ln_b, u16* __restrict__ h_bf) {
  __shared__ __align__(16) u16 xs[16 * DIN];
  __shared__ __align__(16) u16 hsb[16 * DMOD];
  const int t = threadIdx.x;
  const int m0 = blockIdx.x * 16;

  for (int i = t; i < 16 * DIN; i += 256)
    xs[i] = f2bf(x[(size_t)m0 * DIN + i]);
  __syncthreads();

  const int w = t >> 5, L = t & 31, lm = L & 15, half = L >> 4;
  v8f acc[8];
  #pragma unroll
  for (int nt = 0; nt < 8; ++nt) acc[nt] = vzero();

  const u16* abase = xs + lm * DIN + half * 8;
  const u16* wbase = encw_bf + (size_t)(w * 128 + lm) * DIN + half * 8;
  // step i = kt*8 + nt; B addr = wbase + nt*16*DIN + kt*32
  v16bf a  = load_frag(abase);
  v16bf b0 = load_frag(wbase);
  v16bf b1 = load_frag(wbase + 16 * DIN);
  v16bf b2 = load_frag(wbase + 2 * 16 * DIN);
  for (int kt = 0; kt < DIN / 32; ++kt) {
    v16bf an = load_frag(abase + (kt < DIN / 32 - 1 ? kt * 32 + 32 : kt * 32));
    #pragma unroll
    for (int nt = 0; nt < 8; ++nt) {
      int i3 = kt * 8 + nt + 3;
      if (i3 > DIN / 32 * 8 - 1) i3 = DIN / 32 * 8 - 1;
      int nkt = i3 >> 3, nnt = i3 & 7;
      acc[nt] = wmma_bf16(a, b0, acc[nt]);
      b0 = b1; b1 = b2;
      b2 = load_frag(wbase + (size_t)nnt * 16 * DIN + nkt * 32);
    }
    a = an;
  }
  #pragma unroll
  for (int nt = 0; nt < 8; ++nt) {
    int n = w * 128 + nt * 16 + lm;
    float bias = enc_b[n];
    #pragma unroll
    for (int j = 0; j < 8; ++j)
      hsb[(j + half * 8) * DMOD + n] = f2bf(acc[nt][j] + bias);
  }
  __syncthreads();

  const int r = t >> 4, g = t & 15;
  float s = 0.f;
  for (int c = g; c < DMOD; c += 16) s += bf2f(hsb[r * DMOD + c]);
  #pragma unroll
  for (int off = 1; off < 16; off <<= 1) s += __shfl_xor(s, off, 16);
  float mu = s * (1.f / DMOD);
  float vs = 0.f;
  for (int c = g; c < DMOD; c += 16) {
    float d = bf2f(hsb[r * DMOD + c]) - mu; vs += d * d;
  }
  #pragma unroll
  for (int off = 1; off < 16; off <<= 1) vs += __shfl_xor(vs, off, 16);
  float rstd = rsqrtf(vs * (1.f / DMOD) + 1e-5f);
  for (int c = g; c < DMOD; c += 16) {
    float hv = (bf2f(hsb[r * DMOD + c]) - mu) * rstd * ln_g[c] + ln_b[c];
    h_bf[(size_t)(m0 + r) * DMOD + c] = f2bf(hv);
  }
}

// ---------- fused scores -> online softmax -> quantize -> decoder (32 rows/block) ----------
#define SM_HQ 0
#define SM_SV (MT * DMOD * 2)
#define SM_PV (SM_SV + MT * VT * 4)
#define SM_ST (SM_PV + MT * VT * 2)
#define SM_TOTAL (SM_ST + MT * 4 * 4)

__global__ __launch_bounds__(256) void fused_vq(
    const u16* __restrict__ h_bf, const u16* __restrict__ emb_bf,
    const u16* __restrict__ embT_bf, const u16* __restrict__ decw_bf,
    const float* __restrict__ dec_b, float* __restrict__ y_out,
    float* __restrict__ idx_out) {
  __shared__ __align__(16) unsigned char smem[SM_TOTAL];   // ~112.5 KB
  u16*   hq   = (u16*)(smem + SM_HQ);    // 64 KB: h tile, reused as Q tile
  float* Sv   = (float*)(smem + SM_SV);  // 32 KB scores
  u16*   Pv   = (u16*)(smem + SM_PV);    // 16 KB probs (bf16)
  float* mrow = (float*)(smem + SM_ST);
  float* lrow = mrow + MT;
  float* arow = lrow + MT;
  int*   irow = (int*)(arow + MT);

  const int t = threadIdx.x;
  const int m0 = blockIdx.x * MT;
  const int w = t >> 5, L = t & 31, lm = L & 15, half = L >> 4;

  // h tile (64 KB) via Tensor Data Mover, wave 0 issues, all waves sync after.
  if (w == 0) {
    tdm_load_1d(SM_HQ, h_bf + (size_t)m0 * DMOD, (MT * DMOD * 2) / 8);
    __builtin_amdgcn_s_wait_tensorcnt(0);
  }
  if (t < MT) { mrow[t] = -INFINITY; lrow[t] = 0.f; arow[t] = 0.f; irow[t] = 0; }

  v8f qacc0[8], qacc1[8];                          // rows 0-15 / 16-31, 128 D cols per wave
  #pragma unroll
  for (int nt = 0; nt < 8; ++nt) { qacc0[nt] = vzero(); qacc1[nt] = vzero(); }
  __syncthreads();

  const int rrow = t >> 3, g = t & 7;              // 8 threads per row for softmax

  for (int v0 = 0; v0 < VSZ; v0 += VT) {
    // --- scores: wave computes rows 0..31 x cols [v0 + w*32, +32), one 16-col slab at a time ---
    #pragma unroll
    for (int ct = 0; ct < 2; ++ct) {
      v8f s0 = vzero(), s1 = vzero();
      const u16* bbase =
          emb_bf + (size_t)(v0 + w * 32 + ct * 16 + lm) * DMOD + half * 8;
      v16bf b0 = load_frag(bbase);
      v16bf b1 = load_frag(bbase + 32);
      v16bf b2 = load_frag(bbase + 64);
      for (int kt = 0; kt < DMOD / 32; ++kt) {
        int k0 = kt * 32;
        v16bf a0 = load_frag(hq + lm * DMOD + k0 + half * 8);
        v16bf a1 = load_frag(hq + (16 + lm) * DMOD + k0 + half * 8);
        int kp = kt + 3; if (kp > DMOD / 32 - 1) kp = DMOD / 32 - 1;
        s0 = wmma_bf16(a0, b0, s0);
        s1 = wmma_bf16(a1, b0, s1);
        b0 = b1; b1 = b2;
        b2 = load_frag(bbase + kp * 32);
      }
      const float scale = 0.03125f;                // 1/sqrt(1024)
      int c = w * 32 + ct * 16 + lm;
      #pragma unroll
      for (int j = 0; j < 8; ++j) {
        Sv[(j + half * 8) * VT + c]      = s0[j] * scale;
        Sv[(16 + j + half * 8) * VT + c] = s1[j] * scale;
      }
    }
    __syncthreads();

    // --- online softmax bookkeeping: 8 threads per row ---
    {
      float tv = -INFINITY; int ti = 0;
      for (int c = g; c < VT; c += 8) {
        float sv = Sv[rrow * VT + c];
        if (sv > tv) { tv = sv; ti = v0 + c; }
      }
      #pragma unroll
      for (int off = 1; off < 8; off <<= 1) {
        float ov = __shfl_xor(tv, off, 8);
        int   oi = __shfl_xor(ti, off, 8);
        if (ov > tv || (ov == tv && oi < ti)) { tv = ov; ti = oi; }
      }
      float mo = mrow[rrow];
      float mn = fmaxf(mo, tv);
      float al = __expf(mo - mn);
      float ps = 0.f;
      for (int c = g; c < VT; c += 8) {
        float p = __expf(Sv[rrow * VT + c] - mn);
        Pv[rrow * VT + c] = f2bf(p);
        ps += p;
      }
      #pragma unroll
      for (int off = 1; off < 8; off <<= 1) ps += __shfl_xor(ps, off, 8);
      if (g == 0) {
        if (tv > mo) irow[rrow] = ti;
        lrow[rrow] = lrow[rrow] * al + ps;
        mrow[rrow] = mn;
        arow[rrow] = al;
      }
    }
    __syncthreads();

    // --- quantize accumulate: wave owns D cols [w*128, +128) for all 32 rows ---
    #pragma unroll
    for (int j = 0; j < 8; ++j) {
      float a0 = arow[j + half * 8];
      float a1 = arow[16 + j + half * 8];
      #pragma unroll
      for (int nt = 0; nt < 8; ++nt) {
        qacc0[nt][j] = qacc0[nt][j] * a0;
        qacc1[nt][j] = qacc1[nt][j] * a1;
      }
    }
    {
      const u16* bcol = embT_bf + (size_t)(w * 128 + lm) * VSZ + v0 + half * 8;
      // step i = kt*8 + nt; B addr = bcol + nt*16*VSZ + kt*32
      v16bf b0 = load_frag(bcol);
      v16bf b1 = load_frag(bcol + 16 * VSZ);
      v16bf b2 = load_frag(bcol + 2 * 16 * VSZ);
      for (int kt = 0; kt < VT / 32; ++kt) {
        int k0 = kt * 32;
        v16bf a0 = load_frag(Pv + lm * VT + k0 + half * 8);
        v16bf a1 = load_frag(Pv + (16 + lm) * VT + k0 + half * 8);
        #pragma unroll
        for (int nt = 0; nt < 8; ++nt) {
          int i3 = kt * 8 + nt + 3;
          if (i3 > VT / 32 * 8 - 1) i3 = VT / 32 * 8 - 1;
          int nkt = i3 >> 3, nnt = i3 & 7;
          qacc0[nt] = wmma_bf16(a0, b0, qacc0[nt]);
          qacc1[nt] = wmma_bf16(a1, b0, qacc1[nt]);
          b0 = b1; b1 = b2;
          b2 = load_frag(bcol + (size_t)nnt * 16 * VSZ + nkt * 32);
        }
      }
    }
    __syncthreads();
  }

  // --- finalize Q = quantize/l, stage to LDS (reuse hq) ---
  #pragma unroll
  for (int nt = 0; nt < 8; ++nt) {
    int n = w * 128 + nt * 16 + lm;
    #pragma unroll
    for (int j = 0; j < 8; ++j) {
      hq[(j + half * 8) * DMOD + n]      = f2bf(qacc0[nt][j] / lrow[j + half * 8]);
      hq[(16 + j + half * 8) * DMOD + n] = f2bf(qacc1[nt][j] / lrow[16 + j + half * 8]);
    }
  }
  __syncthreads();

  // --- decoder GEMM: wave owns OUT cols [w*64, +64) for all 32 rows ---
  v8f dacc0[4], dacc1[4];
  #pragma unroll
  for (int nt = 0; nt < 4; ++nt) { dacc0[nt] = vzero(); dacc1[nt] = vzero(); }
  {
    const u16* dbase = decw_bf + (size_t)(w * 64 + lm) * DMOD + half * 8;
    // step i = kt*4 + nt; B addr = dbase + nt*16*DMOD + kt*32
    v16bf b0 = load_frag(dbase);
    v16bf b1 = load_frag(dbase + 16 * DMOD);
    v16bf b2 = load_frag(dbase + 2 * 16 * DMOD);
    for (int kt = 0; kt < DMOD / 32; ++kt) {
      int k0 = kt * 32;
      v16bf a0 = load_frag(hq + lm * DMOD + k0 + half * 8);
      v16bf a1 = load_frag(hq + (16 + lm) * DMOD + k0 + half * 8);
      #pragma unroll
      for (int nt = 0; nt < 4; ++nt) {
        int i3 = kt * 4 + nt + 3;
        if (i3 > DMOD / 32 * 4 - 1) i3 = DMOD / 32 * 4 - 1;
        int nkt = i3 >> 2, nnt = i3 & 3;
        dacc0[nt] = wmma_bf16(a0, b0, dacc0[nt]);
        dacc1[nt] = wmma_bf16(a1, b0, dacc1[nt]);
        b0 = b1; b1 = b2;
        b2 = load_frag(dbase + (size_t)nnt * 16 * DMOD + nkt * 32);
      }
    }
  }
  #pragma unroll
  for (int nt = 0; nt < 4; ++nt) {
    int o = w * 64 + nt * 16 + lm;
    float bias = dec_b[o];
    #pragma unroll
    for (int j = 0; j < 8; ++j) {
      y_out[(size_t)(m0 + j + half * 8) * DOUT + o]      = dacc0[nt][j] + bias;
      y_out[(size_t)(m0 + 16 + j + half * 8) * DOUT + o] = dacc1[nt][j] + bias;
    }
  }
  if (t < MT) idx_out[m0 + t] = (float)irow[t];
}

// ---------- host ----------
extern "C" void kernel_launch(void* const* d_in, const int* in_sizes, int n_in,
                              void* d_out, int out_size, void* d_ws, size_t ws_size,
                              hipStream_t stream) {
  const float* x    = (const float*)d_in[0];
  const float* emb  = (const float*)d_in[1];
  const float* encw = (const float*)d_in[2];
  const float* encb = (const float*)d_in[3];
  const float* lng  = (const float*)d_in[4];
  const float* lnb  = (const float*)d_in[5];
  const float* decw = (const float*)d_in[6];
  const float* decb = (const float*)d_in[7];
  float* out = (float*)d_out;

  char* p = (char*)d_ws;
  u16* encw_bf = (u16*)p; p += (size_t)DMOD * DIN * 2;
  u16* decw_bf = (u16*)p; p += (size_t)DOUT * DMOD * 2;
  u16* emb_bf  = (u16*)p; p += (size_t)VSZ * DMOD * 2;
  u16* embT_bf = (u16*)p; p += (size_t)VSZ * DMOD * 2;
  u16* h_bf    = (u16*)p; p += (size_t)MTOT * DMOD * 2;

  prep_w<<<512, 256, 0, stream>>>(encw, decw, encw_bf, decw_bf, DMOD * DIN);
  prep_emb<<<dim3(DMOD / 32, VSZ / 32), dim3(32, 8), 0, stream>>>(emb, emb_bf, embT_bf);
  encode_ln<<<MTOT / 16, 256, 0, stream>>>(x, encw_bf, encb, lng, lnb, h_bf);
  fused_vq<<<MTOT / MT, 256, 0, stream>>>(h_bf, emb_bf, embT_bf, decw_bf, decb,
                                          out, out + (size_t)MTOT * DOUT);
}
